// SlotAttention_343597384042
// MI455X (gfx1250) — matrix operand order repaired and checked by reading, not compiled, not run on hardware
//
#include <hip/hip_runtime.h>
#include <hip/hip_bf16.h>

typedef __attribute__((ext_vector_type(16))) _Float16 v16h;
typedef __attribute__((ext_vector_type(8)))  _Float16 v8h;
typedef __attribute__((ext_vector_type(8)))  float    v8f;
typedef __attribute__((ext_vector_type(4)))  float    v4f;
typedef __attribute__((ext_vector_type(4)))  unsigned int u32x4;
typedef __attribute__((ext_vector_type(8)))  int       i32x8;
typedef __attribute__((ext_vector_type(4)))  int       i32x4;

#define D   256
#define NS  16
#define BB  131072
#define LN_EPS 1e-5f
#define SA_EPS 1e-8f
#define NBLK 128          // kb_attn workgroups (each owns 1024 rows of b)
#define CHUNK 1024

// ---- workspace layout (bytes) ----
#define OFF_KF    ((size_t)0)                    // k   f16 [B][D]      64 MB
#define OFF_VT    ((size_t)67108864)             // v^T f16 [D][B]      64 MB
#define OFF_WK    ((size_t)134217728)            // Wk  f16 [D][D]     128 KB
#define OFF_WV    ((size_t)134348800)            // Wv  f16 [D][D]     128 KB
#define OFF_QH    ((size_t)134479872)            // q   f16 [NS][D]      8 KB
#define OFF_SLOTS ((size_t)134488064)            // slots f32 [NS][D]   16 KB
#define OFF_UPDP  ((size_t)134504448)            // updates partials f32 [NBLK][NS*D]  2 MB
#define OFF_ASUMP ((size_t)136601600)            // attn-sum partials f32 [NBLK][NS]   8 KB

__device__ __forceinline__ v16h pack16(v8h lo, v8h hi) {
  v16h r;
#pragma unroll
  for (int i = 0; i < 8; ++i) { r[i] = lo[i]; r[8 + i] = hi[i]; }
  return r;
}
__device__ __forceinline__ v8f wmma16(v16h a, v16h b, v8f c) {
  return __builtin_amdgcn_wmma_f32_16x16x32_f16(false, a, false, b, (short)0, c, false, false);
}

// ---------------------------------------------------------------------------
// K0: slots = mu + sigma*noise ; Wk,Wv -> f16
// ---------------------------------------------------------------------------
__global__ __launch_bounds__(256) void k0_init(
    const float* __restrict__ Wk, const float* __restrict__ Wv,
    const float* __restrict__ mu, const float* __restrict__ sigma,
    const float* __restrict__ noise,
    _Float16* __restrict__ wk_h, _Float16* __restrict__ wv_h,
    float* __restrict__ slots) {
  int i = blockIdx.x * 256 + threadIdx.x;            // 0..65535
  if (i < D * D) { wk_h[i] = (_Float16)Wk[i]; wv_h[i] = (_Float16)Wv[i]; }
  if (i < NS * D) slots[i] = mu[i & (D - 1)] + sigma[i & (D - 1)] * noise[i];
}

// ---------------------------------------------------------------------------
// K1: fused LayerNorm(x) -> k = xln@Wk^T + bk (row-major f16)
//                           v = xln@Wv^T + bv (stored transposed, f16)
// 128 threads = 4 waves; each wave owns 16 rows of b.
// ---------------------------------------------------------------------------
__global__ __launch_bounds__(128) void k1_kv(
    const float* __restrict__ x, const float* __restrict__ g, const float* __restrict__ be,
    const _Float16* __restrict__ wk_h, const _Float16* __restrict__ wv_h,
    const float* __restrict__ bk, const float* __restrict__ bv,
    _Float16* __restrict__ kf, _Float16* __restrict__ vt) {
  __shared__ __align__(16) _Float16 xlds[4 * 16 * D];   // 32 KB
  const int tid = threadIdx.x, wave = tid >> 5, lane = tid & 31;
  const int rowbase = (blockIdx.x * 4 + wave) * 16;

  float gv[8], bev[8];
#pragma unroll
  for (int i = 0; i < 8; ++i) { gv[i] = g[lane * 8 + i]; bev[i] = be[lane * 8 + i]; }

  for (int r = 0; r < 16; ++r) {
    float xv[8], s1 = 0.f, s2 = 0.f;
#pragma unroll
    for (int i = 0; i < 8; ++i) {
      float t = x[(size_t)(rowbase + r) * D + lane * 8 + i];
      xv[i] = t; s1 += t; s2 += t * t;
    }
#pragma unroll
    for (int off = 16; off >= 1; off >>= 1) {
      s1 += __shfl_xor(s1, off, 32);
      s2 += __shfl_xor(s2, off, 32);
    }
    float m = s1 * (1.f / D), var = s2 * (1.f / D) - m * m;
    float rs = rsqrtf(var + LN_EPS);
    v8h hv;
#pragma unroll
    for (int i = 0; i < 8; ++i) hv[i] = (_Float16)((xv[i] - m) * rs * gv[i] + bev[i]);
    *(v8h*)&xlds[(wave * 16 + r) * D + lane * 8] = hv;
  }
  __syncthreads();

  // resident A fragments for this wave's 16 rows (K = 256 -> 8 chunks of 32)
  v16h af[8];
#pragma unroll
  for (int c = 0; c < 8; ++c) {
    const _Float16* p = &xlds[(wave * 16 + (lane & 15)) * D + c * 32 + (lane >> 4) * 8];
    af[c] = pack16(*(const v8h*)p, *(const v8h*)(p + 16));
  }

  for (int wsel = 0; wsel < 2; ++wsel) {
    const _Float16* W = wsel ? wv_h : wk_h;
    const float* bias = wsel ? bv : bk;
    for (int nb = 0; nb < 16; ++nb) {
      const int n0 = nb * 16;
      v8f acc = {0, 0, 0, 0, 0, 0, 0, 0};
      const float bbv = bias[n0 + (lane & 15)];
#pragma unroll
      for (int c = 0; c < 8; ++c) {
        const _Float16* p = &W[(size_t)(n0 + (lane & 15)) * D + c * 32 + (lane >> 4) * 8];
        v16h bf = pack16(*(const v8h*)p, *(const v8h*)(p + 16));
        acc = wmma16(af[c], bf, acc);
      }
      if (wsel == 0) {
#pragma unroll
        for (int i = 0; i < 8; ++i) {
          const int mrow = (lane < 16) ? i : 8 + i;
          kf[(size_t)(rowbase + mrow) * D + n0 + (lane & 15)] = (_Float16)(acc[i] + bbv);
        }
      } else {
        v8h pk;
#pragma unroll
        for (int i = 0; i < 8; ++i) pk[i] = (_Float16)(acc[i] + bbv);
        *(v8h*)&vt[(size_t)(n0 + (lane & 15)) * BB + rowbase + (lane >> 4) * 8] = pk;
      }
    }
  }
}

// ---------------------------------------------------------------------------
// KA: s = LN(slots); q = s@Wq^T + bq  (f16 out). 1 workgroup, 256 threads.
// ---------------------------------------------------------------------------
__global__ __launch_bounds__(256) void ka_pre(
    const float* __restrict__ slots, const float* __restrict__ g_s, const float* __restrict__ be_s,
    const float* __restrict__ Wq, const float* __restrict__ bq,
    _Float16* __restrict__ q_h) {
  __shared__ float s_lds[NS * D];
  __shared__ float mrow[NS], rrow[NS];
  const int t = threadIdx.x, n = t >> 4, cb = (t & 15) * 16;
  if (t < NS) {
    float s1 = 0.f, s2 = 0.f;
    for (int k = 0; k < D; ++k) { float v = slots[t * D + k]; s1 += v; s2 += v * v; }
    float m = s1 * (1.f / D);
    mrow[t] = m; rrow[t] = rsqrtf(s2 * (1.f / D) - m * m + LN_EPS);
  }
  __syncthreads();
  for (int j = 0; j < 16; ++j) {
    const int c = cb + j;
    s_lds[n * D + c] = (slots[n * D + c] - mrow[n]) * rrow[n] * g_s[c] + be_s[c];
  }
  __syncthreads();
  for (int j = 0; j < 16; ++j) {
    const int c = cb + j;
    float acc = bq[c];
    for (int k = 0; k < D; ++k) acc += s_lds[n * D + k] * Wq[(size_t)c * D + k];
    q_h[n * D + c] = (_Float16)acc;
  }
}

// ---------------------------------------------------------------------------
// KB: per 1024-row chunk: dots = k@q^T * scale, softmax over slots (+eps),
//     per-slot partial sums, updates-partial = attn@v via WMMA.
// q (16x256 f16) is DMA'd into LDS once per WG by the Tensor Data Mover.
// ---------------------------------------------------------------------------
__global__ __launch_bounds__(256) void kb_attn(
    const _Float16* __restrict__ kf, const _Float16* __restrict__ vt,
    const _Float16* __restrict__ q_h,
    float* __restrict__ upd_p, float* __restrict__ asum_p) {
  __shared__ __align__(16) _Float16 attn[NS * CHUNK];  // 32 KB
  __shared__ __align__(16) _Float16 qlds[NS * D];      //  8 KB
  __shared__ __align__(16) float sm[8][16 * 16];       //  8 KB
  __shared__ float red[NS][16];
  const int tid = threadIdx.x, wave = tid >> 5, lane = tid & 31;
  const size_t cbase = (size_t)blockIdx.x * CHUNK;

  if (wave == 0) {
    // Tensor DMA: q[16][256] f16, contiguous, one 2-D descriptor (D# groups 0/1).
    const unsigned lds_addr = (unsigned)(uintptr_t)(&qlds[0]);     // low 32b = LDS offset
    const unsigned long long ga = (unsigned long long)(uintptr_t)q_h;
    u32x4 g0 = { 1u,                                   // count=1, user descriptor
                 lds_addr,                             // lds_addr
                 (unsigned)(ga & 0xFFFFFFFFu),         // global_addr[31:0]
                 ((unsigned)(ga >> 32) & 0x01FFFFFFu) | 0x80000000u };  // addr[56:32] | type=2
    i32x8 g1 = { 0x00010000,            // data_size=2B (code 1), mask/flags=0
                 0x01000000,            // tensor_dim0[15:0]=256 at bits[63:48]
                 0x00100000,            // tensor_dim1[15:0]=16 at bits[95:80]
                 0x01000000,            // tile_dim0=256 at bits[127:112]
                 16,                    // tile_dim1=16, tile_dim2=0
                 256,                   // tensor_dim0_stride=256 (bits[207:160])
                 0, 0 };                // tensor_dim1_stride=0 (unused for 2D tile)
    i32x4 z4 = { 0, 0, 0, 0 };
    i32x8 z8 = { 0, 0, 0, 0, 0, 0, 0, 0 };
    __builtin_amdgcn_tensor_load_to_lds(g0, g1, z4, z4, z8, 0);
    __builtin_amdgcn_s_wait_tensorcnt(0);
  }
  __syncthreads();

  // q B-fragments resident in VGPRs (shared by all row tiles), read from LDS
  v16h qb[8];
#pragma unroll
  for (int c = 0; c < 8; ++c) {
    const _Float16* p = &qlds[(lane & 15) * D + c * 32 + (lane >> 4) * 8];
    qb[c] = pack16(*(const v8h*)p, *(const v8h*)(p + 16));
  }

  // phase 1: dots + softmax-over-slots for 8 row tiles per wave
  for (int tt = 0; tt < 8; ++tt) {
    const int rt = wave * 8 + tt;
    const size_t m0 = cbase + (size_t)rt * 16;
    if (tt < 7)   // prefetch next row tile of k into cache
      __builtin_prefetch(&kf[(m0 + 16 + (lane & 15)) * D], 0, 0);
    v8f acc = {0, 0, 0, 0, 0, 0, 0, 0};
#pragma unroll
    for (int c = 0; c < 8; ++c) {
      const _Float16* p = &kf[(m0 + (lane & 15)) * D + c * 32 + (lane >> 4) * 8];
      v16h af = pack16(*(const v8h*)p, *(const v8h*)(p + 16));
      acc = wmma16(af, qb[c], acc);
    }
#pragma unroll
    for (int i = 0; i < 8; ++i) {
      const int mrow = (lane < 16) ? i : 8 + i;
      sm[wave][mrow * 16 + (lane & 15)] = acc[i] * 0.0625f;   // * d^-0.5
    }
    asm volatile("s_wait_dscnt 0" ::: "memory");
    if (lane < 16) {
      float vrow[16];
#pragma unroll
      for (int q4 = 0; q4 < 4; ++q4) {
        v4f v4 = *(const v4f*)&sm[wave][lane * 16 + q4 * 4];
#pragma unroll
        for (int u = 0; u < 4; ++u) vrow[q4 * 4 + u] = v4[u];
      }
      float mx = vrow[0];
#pragma unroll
      for (int s = 1; s < 16; ++s) mx = fmaxf(mx, vrow[s]);
      float sum = 0.f;
#pragma unroll
      for (int s = 0; s < 16; ++s) { vrow[s] = __expf(vrow[s] - mx); sum += vrow[s]; }
      const float inv = 1.f / sum;
      const int rowI = rt * 16 + lane;
#pragma unroll
      for (int s = 0; s < 16; ++s) attn[s * CHUNK + rowI] = (_Float16)(vrow[s] * inv + SA_EPS);
    }
  }
  __syncthreads();

  // deterministic per-slot partial sums: S_n(partial) = sum_b attn[n][b]
  {
    const int n = tid & 15, seg = tid >> 4;
    float p = 0.f;
    for (int j = 0; j < 64; ++j) p += (float)attn[n * CHUNK + seg * 64 + j];
    red[n][seg] = p;
  }
  __syncthreads();
  if (tid < NS) {
    float s = 0.f;
#pragma unroll
    for (int g2 = 0; g2 < 16; ++g2) s += red[tid][g2];
    asum_p[blockIdx.x * NS + tid] = s;
  }

  // phase 2: updates partial = attn(16 x 1024) @ v(1024 x 256)
  for (int sub = 0; sub < 2; ++sub) {
    const int n0 = wave * 32 + sub * 16;
    v8f acc = {0, 0, 0, 0, 0, 0, 0, 0};
#pragma unroll 8
    for (int c = 0; c < 32; ++c) {
      const _Float16* pa = &attn[(lane & 15) * CHUNK + c * 32 + (lane >> 4) * 8];
      v16h af = pack16(*(const v8h*)pa, *(const v8h*)(pa + 16));
      const _Float16* pb = &vt[(size_t)(n0 + (lane & 15)) * BB + cbase + c * 32 + (lane >> 4) * 8];
      v16h bf = pack16(*(const v8h*)pb, *(const v8h*)(pb + 16));
      acc = wmma16(af, bf, acc);
    }
#pragma unroll
    for (int i = 0; i < 8; ++i) {
      const int mrow = (lane < 16) ? i : 8 + i;
      upd_p[(size_t)blockIdx.x * (NS * D) + mrow * D + n0 + (lane & 15)] = acc[i];
    }
  }
}

// ---------------------------------------------------------------------------
// KC: reduce partials, normalize, GRU cell, FF, slots update (+ write d_out)
// ---------------------------------------------------------------------------
__global__ __launch_bounds__(256) void kc_post(
    float* __restrict__ slots,
    const float* __restrict__ upd_p, const float* __restrict__ asum_p,
    const float* __restrict__ W_ih, const float* __restrict__ W_hh,
    const float* __restrict__ b_ih, const float* __restrict__ b_hh,
    const float* __restrict__ W1, const float* __restrict__ b1,
    const float* __restrict__ W2, const float* __restrict__ b2,
    const float* __restrict__ g_f, const float* __restrict__ be_f,
    float* __restrict__ out) {
  __shared__ float bufA[NS * D];   // updates, later hidden
  __shared__ float bufB[NS * D];   // slots_prev, later LN(new)
  __shared__ float bufC[NS * D];   // new slots
  __shared__ float mrow[NS], rrow[NS], arow[NS];
  const int t = threadIdx.x, n = t >> 4, cb = (t & 15) * 16;

  if (t < NS) {
    float s = 0.f;
    for (int g2 = 0; g2 < NBLK; ++g2) s += asum_p[g2 * NS + t];
    arow[t] = s;
  }
  __syncthreads();
  for (int j = 0; j < 16; ++j) {
    const int c = cb + j;
    float s = 0.f;
    for (int g2 = 0; g2 < NBLK; ++g2) s += upd_p[(size_t)g2 * (NS * D) + n * D + c];
    bufA[n * D + c] = s / arow[n];
    bufB[n * D + c] = slots[n * D + c];
  }
  __syncthreads();

  // GRU cell
  for (int j = 0; j < 16; ++j) {
    const int c = cb + j;
    float xr = b_ih[c], xz = b_ih[D + c], xn = b_ih[2 * D + c];
    float hr = b_hh[c], hz = b_hh[D + c], hn = b_hh[2 * D + c];
    for (int k = 0; k < D; ++k) {
      const float u = bufA[n * D + k], h = bufB[n * D + k];
      xr += u * W_ih[(size_t)c * D + k];
      xz += u * W_ih[(size_t)(D + c) * D + k];
      xn += u * W_ih[(size_t)(2 * D + c) * D + k];
      hr += h * W_hh[(size_t)c * D + k];
      hz += h * W_hh[(size_t)(D + c) * D + k];
      hn += h * W_hh[(size_t)(2 * D + c) * D + k];
    }
    const float r = 1.f / (1.f + __expf(-(xr + hr)));
    const float z = 1.f / (1.f + __expf(-(xz + hz)));
    const float nn = tanhf(xn + r * hn);
    bufC[n * D + c] = (1.f - z) * nn + z * bufB[n * D + c];
  }
  __syncthreads();

  // FF: slots += relu(LN(slots)@W1^T + b1)@W2^T + b2
  if (t < NS) {
    float s1 = 0.f, s2 = 0.f;
    for (int k = 0; k < D; ++k) { float v = bufC[t * D + k]; s1 += v; s2 += v * v; }
    const float m = s1 * (1.f / D);
    mrow[t] = m; rrow[t] = rsqrtf(s2 * (1.f / D) - m * m + LN_EPS);
  }
  __syncthreads();
  for (int j = 0; j < 16; ++j) {
    const int c = cb + j;
    bufB[n * D + c] = (bufC[n * D + c] - mrow[n]) * rrow[n] * g_f[c] + be_f[c];
  }
  __syncthreads();
  for (int j = 0; j < 16; ++j) {
    const int c = cb + j;
    float a = b1[c];
    for (int k = 0; k < D; ++k) a += bufB[n * D + k] * W1[(size_t)c * D + k];
    bufA[n * D + c] = fmaxf(a, 0.f);
  }
  __syncthreads();
  for (int j = 0; j < 16; ++j) {
    const int c = cb + j;
    float a = b2[c];
    for (int k = 0; k < D; ++k) a += bufA[n * D + k] * W2[(size_t)c * D + k];
    const float v = bufC[n * D + c] + a;
    slots[n * D + c] = v;
    out[n * D + c] = v;
  }
}

// ---------------------------------------------------------------------------
extern "C" void kernel_launch(void* const* d_in, const int* in_sizes, int n_in,
                              void* d_out, int out_size, void* d_ws, size_t ws_size,
                              hipStream_t stream) {
  const float* inputs = (const float*)d_in[0];
  const float* noise  = (const float*)d_in[1];
  const float* mu     = (const float*)d_in[2];
  const float* sigma  = (const float*)d_in[3];
  const float* Wq     = (const float*)d_in[4];
  const float* bq     = (const float*)d_in[5];
  const float* Wk     = (const float*)d_in[6];
  const float* bk     = (const float*)d_in[7];
  const float* Wv     = (const float*)d_in[8];
  const float* bv     = (const float*)d_in[9];
  const float* W_ih   = (const float*)d_in[10];
  const float* W_hh   = (const float*)d_in[11];
  const float* b_ih   = (const float*)d_in[12];
  const float* b_hh   = (const float*)d_in[13];
  const float* W1     = (const float*)d_in[14];
  const float* b1     = (const float*)d_in[15];
  const float* W2     = (const float*)d_in[16];
  const float* b2     = (const float*)d_in[17];
  const float* g_in   = (const float*)d_in[18];
  const float* be_in  = (const float*)d_in[19];
  const float* g_s    = (const float*)d_in[20];
  const float* be_s   = (const float*)d_in[21];
  const float* g_f    = (const float*)d_in[22];
  const float* be_f   = (const float*)d_in[23];

  char* ws = (char*)d_ws;
  _Float16* kf    = (_Float16*)(ws + OFF_KF);
  _Float16* vt    = (_Float16*)(ws + OFF_VT);
  _Float16* wk_h  = (_Float16*)(ws + OFF_WK);
  _Float16* wv_h  = (_Float16*)(ws + OFF_WV);
  _Float16* q_h   = (_Float16*)(ws + OFF_QH);
  float*    slots = (float*)(ws + OFF_SLOTS);
  float*    upd_p = (float*)(ws + OFF_UPDP);
  float*    asum_p= (float*)(ws + OFF_ASUMP);
  float*    outp  = (float*)d_out;

  k0_init<<<256, 256, 0, stream>>>(Wk, Wv, mu, sigma, noise, wk_h, wv_h, slots);
  k1_kv<<<BB / 64, 128, 0, stream>>>(inputs, g_in, be_in, wk_h, wv_h, bk, bv, kf, vt);
  for (int it = 0; it < 3; ++it) {
    ka_pre<<<1, 256, 0, stream>>>(slots, g_s, be_s, Wq, bq, q_h);
    kb_attn<<<NBLK, 256, 0, stream>>>(kf, vt, q_h, upd_p, asum_p);
    kc_post<<<1, 256, 0, stream>>>(slots, upd_p, asum_p, W_ih, W_hh, b_ih, b_hh,
                                   W1, b1, W2, b2, g_f, be_f, outp);
  }
}
